// Mamba1_72473278152755
// MI455X (gfx1250) — matrix operand order repaired and checked by reading, not compiled, not run on hardware
//
#include <hip/hip_runtime.h>
#include <hip/hip_bf16.h>

// ---------------- problem constants (from reference) ----------------
#define B_    2
#define L_    2048
#define DM    768
#define IN_   1536          // EXPAND * D_MODEL
#define NS    16            // d_state
#define MTOT  (B_ * L_)     // 4096 rows of activations

// ---------------- types ----------------
typedef __attribute__((ext_vector_type(16))) __bf16          v16bf;
typedef __attribute__((ext_vector_type(8)))  float           v8f;
typedef __attribute__((ext_vector_type(8)))  unsigned short  u16x8;
typedef __attribute__((ext_vector_type(4)))  unsigned int    u32x4;
typedef __attribute__((ext_vector_type(8)))  int             i32x8;
typedef __attribute__((ext_vector_type(4)))  int             i32x4;

union FragBF { u16x8 h[2]; v16bf bf; };

__device__ __forceinline__ unsigned short f32_to_bf16(float f) {
    union { float f; unsigned u; } v; v.f = f;
    unsigned u = v.u;
    unsigned r = u + 0x7FFFu + ((u >> 16) & 1u);   // round-to-nearest-even
    return (unsigned short)(r >> 16);
}

__device__ __forceinline__ float silu_f(float x) {
    return x * (1.0f / (1.0f + expf(-x)));
}

__device__ __forceinline__ unsigned lds_lo32(const void* p) {
    // generic pointer to LDS: addr[31:0] is the LDS offset (aperture rules)
    return (unsigned)(unsigned long long)(uintptr_t)p;
}

// ---------------- Tensor Data Mover: 2D tile (rows x tileK bf16) -> LDS ----------------
// LDS destination rows padded: pad_interval = 64B (16 dwords), pad_amount = 16B
// (4 dwords) => 80B row pitch == 40 ushorts, matching the fragment layout below.
__device__ __forceinline__ void tdm_load_tile_2d(unsigned lds_addr,
                                                 const unsigned short* gptr,
                                                 int strideK, int rows, int tileK) {
    unsigned long long ga = (unsigned long long)(uintptr_t)gptr;
    u32x4 g0;
    g0[0] = 1u;                                            // count=1 (valid), user mode
    g0[1] = lds_addr;                                      // lds_addr [63:32]
    g0[2] = (unsigned)(ga & 0xFFFFFFFFu);                  // global_addr [95:64]
    g0[3] = (unsigned)((ga >> 32) & 0x01FFFFFFu)           // global_addr [120:96]
          | (2u << 30);                                    // type=2 ("image") [127:126]
    i32x8 g1;
    g1[0] = (1 << 16)        // data_size = 2 bytes
          | (1 << 20)        // pad_enable
          | (3 << 22)        // pad_interval: 16 dwords (64 B)
          | (3 << 25);       // pad_amount : 4 dwords (16 B)
    g1[1] = (strideK & 0xFFFF) << 16;     // tensor_dim0[15:0]
    g1[2] = (rows & 0xFFFF) << 16;        // tensor_dim0[31:16]=0 | tensor_dim1[15:0]
    g1[3] = (tileK & 0xFFFF) << 16;       // tensor_dim1[31:16]=0 | tile_dim0
    g1[4] = (rows & 0xFFFF);              // tile_dim1 | tile_dim2=0
    g1[5] = strideK;                      // tensor_dim0_stride[31:0]
    g1[6] = 0;                            // stride hi | tensor_dim1_stride lo
    g1[7] = 0;
    i32x4 gz4;
    gz4[0] = 0; gz4[1] = 0; gz4[2] = 0; gz4[3] = 0;        // 2D: dims 2..4 unused
    i32x8 gz8;
#pragma unroll
    for (int i = 0; i < 8; ++i) gz8[i] = 0;
    // 6-arg toolchain form: (g0, g1, g2, g3, extra, cpol)
    __builtin_amdgcn_tensor_load_to_lds(g0, g1, gz4, gz4, gz8, 0);
}

// ---------------- f32 -> bf16 cast ----------------
__global__ void cast_f32_bf16_kernel(const float* __restrict__ src,
                                     unsigned short* __restrict__ dst, int n) {
    int i = blockIdx.x * blockDim.x + threadIdx.x;
    if (i < n) dst[i] = f32_to_bf16(src[i]);
}

// ---------------- TN GEMM: C[m,n] = sum_k A[m,k] * B[n,k] ----------------
// A: M x K bf16 row-major, B: N x K bf16 row-major, C: M x ldc f32 row-major.
// Block tile 128(M) x 128(N); grid = (N/128, M/128); 256 threads = 8 waves
// arranged 4(M) x 2(N). Wave tile 32 x 64: 2 A frags x 4 B frags -> 8 WMMAs.
// Tile staging is done by the Tensor Data Mover (double buffered, TENSORcnt),
// so the vector pipe only does fragment ds_loads + WMMAs.
__global__ __launch_bounds__(256)
void gemm_bf16_tn_kernel(const unsigned short* __restrict__ A,
                         const unsigned short* __restrict__ Bm,
                         float* __restrict__ C,
                         int K, int ldc) {
    __shared__ unsigned short lA[2][128 * 40];   // 128 x 32 bf16, rows padded to 40
    __shared__ unsigned short lB[2][128 * 40];

    const int tid  = threadIdx.x;
    const int lane = tid & 31;
    const int wave = tid >> 5;
    const int wm   = wave >> 1;        // 0..3  (M sub-tile, 32 rows each)
    const int wn   = wave & 1;         // 0..1  (N sub-tile, 64 cols each)

    const int mBlk = blockIdx.y * 128;
    const int nBlk = blockIdx.x * 128;

    const unsigned short* gAblk = A  + (size_t)mBlk * K;   // tile origin (k advances)
    const unsigned short* gBblk = Bm + (size_t)nBlk * K;

    const unsigned ldsA0 = lds_lo32(&lA[0][0]);
    const unsigned ldsA1 = lds_lo32(&lA[1][0]);
    const unsigned ldsB0 = lds_lo32(&lB[0][0]);
    const unsigned ldsB1 = lds_lo32(&lB[1][0]);

    const int nk = K >> 5;   // K / 32

    // ---- prologue: TDM stage 0 ----
    if (wave == 0) {
        tdm_load_tile_2d(ldsA0, gAblk, K, 128, 32);
        tdm_load_tile_2d(ldsB0, gBblk, K, 128, 32);
        __builtin_amdgcn_s_wait_tensorcnt(0);
    }
    __syncthreads();

    v8f acc[8];
#pragma unroll
    for (int j = 0; j < 8; ++j)
#pragma unroll
        for (int i = 0; i < 8; ++i) acc[j][i] = 0.f;

    // ISA 16-bit fragment lane layout:
    //  A 16x32: lane holds row M=lane&15; K chunks at (lane>>4)*8 and +16
    //  B 32x16: lane holds col N=lane&15; K chunks at (lane>>4)*8 and +16
    const int aRow0 = wm * 32 + (lane & 15);
    const int aK0   = (lane >> 4) * 8;
    const int bK0   = (lane >> 4) * 8;

    for (int kt = 0; kt < nk; ++kt) {
        const int cur = kt & 1;

        // ---- issue TDM for the next k-tile into the other buffer ----
        if (wave == 0 && kt + 1 < nk) {
            const size_t koff = (size_t)(kt + 1) * 32;
            tdm_load_tile_2d(cur ? ldsA0 : ldsA1, gAblk + koff, K, 128, 32);
            tdm_load_tile_2d(cur ? ldsB0 : ldsB1, gBblk + koff, K, 128, 32);
            if (kt + 2 < nk) {
                __builtin_prefetch(gAblk + (size_t)(kt + 2) * 32, 0, 1);
                __builtin_prefetch(gBblk + (size_t)(kt + 2) * 32, 0, 1);
            }
        }

        // ---- hoist ALL fragment loads, then issue 8 WMMAs back-to-back ----
        FragBF fa[2], fb[4];
#pragma unroll
        for (int s = 0; s < 2; ++s) {
            const int r = (aRow0 + s * 16) * 40 + aK0;
            fa[s].h[0] = *(const u16x8*)&lA[cur][r];
            fa[s].h[1] = *(const u16x8*)&lA[cur][r + 16];
        }
#pragma unroll
        for (int j = 0; j < 4; ++j) {
            const int r = (wn * 64 + j * 16 + (lane & 15)) * 40 + bK0;
            fb[j].h[0] = *(const u16x8*)&lB[cur][r];
            fb[j].h[1] = *(const u16x8*)&lB[cur][r + 16];
        }
#pragma unroll
        for (int s = 0; s < 2; ++s)
#pragma unroll
            for (int j = 0; j < 4; ++j)
                acc[s * 4 + j] = __builtin_amdgcn_wmma_f32_16x16x32_bf16(
                    false, fa[s].bf, false, fb[j].bf,
                    (short)0, acc[s * 4 + j], false, false);

        // ---- publish next buffer: TDM completion, then WG barrier ----
        if (wave == 0 && kt + 1 < nk)
            __builtin_amdgcn_s_wait_tensorcnt(0);
        __syncthreads();
    }

    // C/D f32 16x16 layout: lanes 0-15 -> N=lane, M=v; lanes 16-31 -> N=lane-16, M=8+v
    const int sRow = (lane >> 4) * 8;
    const int sCol = lane & 15;
#pragma unroll
    for (int s = 0; s < 2; ++s)
#pragma unroll
        for (int j = 0; j < 4; ++j) {
            float* cp = C + (size_t)(mBlk + wm * 32 + s * 16 + sRow) * ldc
                          + (nBlk + wn * 64 + j * 16 + sCol);
#pragma unroll
            for (int v = 0; v < 8; ++v) cp[(size_t)v * ldc] = acc[s * 4 + j][v];
        }
}

// ---------------- depthwise causal conv (k=4) + bias + SiLU ----------------
// xz: (MTOT, 2*IN_) f32, proj = cols [0, IN_). Writes u (f32 and bf16), (MTOT, IN_).
__global__ void conv_silu_kernel(const float* __restrict__ xz,
                                 const float* __restrict__ cw,
                                 const float* __restrict__ cb,
                                 float* __restrict__ u32,
                                 unsigned short* __restrict__ ubf) {
    int i = blockIdx.x * blockDim.x + threadIdx.x;
    if (i >= MTOT * IN_) return;
    int d = i % IN_;
    int m = i / IN_;
    int l = m % L_;

    const ptrdiff_t st = 2 * IN_;
    const float* base = xz + (size_t)m * st + d;

    float x0 = (l >= 3) ? base[-3 * st] : 0.f;
    float x1 = (l >= 2) ? base[-2 * st] : 0.f;
    float x2 = (l >= 1) ? base[-1 * st] : 0.f;
    float x3 = base[0];

    float acc = cb[d] + cw[d*4+0]*x0 + cw[d*4+1]*x1 + cw[d*4+2]*x2 + cw[d*4+3]*x3;
    float s = silu_f(acc);
    u32[i] = s;
    ubf[i] = f32_to_bf16(s);
}

// ---------------- dt = softplus(delta + dt_proj_b + dt_bias), in place ----------------
__global__ void dt_softplus_kernel(float* __restrict__ delta,
                                   const float* __restrict__ dpb,
                                   const float* __restrict__ dtb) {
    int i = blockIdx.x * blockDim.x + threadIdx.x;
    if (i >= MTOT * IN_) return;
    int d = i % IN_;
    float x = delta[i] + dpb[d] + dtb[d];
    delta[i] = (x > 20.f) ? x : log1pf(expf(x));
}

// ---------------- selective scan + epilogue ----------------
// One lane per (b, d, n); 16-lane xor-shuffle reduction over n for the C dot.
// Fuses y = (scan + D*u) * silu(z); writes bf16 for out_proj GEMM.
__global__ __launch_bounds__(256)
void scan_kernel(const float* __restrict__ dt,     // (MTOT, IN_)
                 const float* __restrict__ u32,    // (MTOT, IN_)
                 const float* __restrict__ xz,     // (MTOT, 2*IN_), gate at IN_+d
                 const float* __restrict__ A_log,  // (IN_, NS)
                 const float* __restrict__ Bp,     // (IN_, NS)
                 const float* __restrict__ Cp,     // (IN_, NS)
                 const float* __restrict__ Dp,     // (IN_,)
                 unsigned short* __restrict__ ybf) // (MTOT, IN_)
{
    int gid = blockIdx.x * blockDim.x + threadIdx.x;  // B_*IN_*NS lanes
    int n = gid & 15;
    int g = gid >> 4;
    int d = g % IN_;
    int b = g / IN_;

    const float a   = -expf(A_log[d * NS + n]);
    const float Bdn = Bp[d * NS + n];
    const float Cdn = Cp[d * NS + n];
    const float Dd  = Dp[d];

    float h = 0.f;
    const size_t rowBase = (size_t)b * L_;
    for (int l = 0; l < L_; ++l) {
        const size_t m = rowBase + l;
        const float dtv = dt[m * IN_ + d];
        const float uv  = u32[m * IN_ + d];
        h = expf(dtv * a) * h + (dtv * uv) * Bdn;
        float p = h * Cdn;
        p += __shfl_xor(p, 1, 16);
        p += __shfl_xor(p, 2, 16);
        p += __shfl_xor(p, 4, 16);
        p += __shfl_xor(p, 8, 16);
        if (n == 0) {
            const float z = xz[m * (2 * IN_) + IN_ + d];
            const float y = (p + Dd * uv) * silu_f(z);
            ybf[m * IN_ + d] = f32_to_bf16(y);
        }
    }
}

// ---------------- host orchestration ----------------
extern "C" void kernel_launch(void* const* d_in, const int* in_sizes, int n_in,
                              void* d_out, int out_size, void* d_ws, size_t ws_size,
                              hipStream_t stream) {
    const float* hs   = (const float*)d_in[0];   // (B, L, DM)
    const float* wi   = (const float*)d_in[1];   // (2*IN_, DM)
    const float* wo   = (const float*)d_in[2];   // (DM, IN_)
    const float* wd   = (const float*)d_in[3];   // (IN_, IN_)
    const float* dpb  = (const float*)d_in[4];   // (IN_,)
    const float* cw   = (const float*)d_in[5];   // (IN_, 4)
    const float* cb   = (const float*)d_in[6];   // (IN_,)
    const float* alog = (const float*)d_in[7];   // (IN_, NS)
    const float* Bp   = (const float*)d_in[8];   // (IN_, NS)
    const float* Cp   = (const float*)d_in[9];   // (IN_, NS)
    const float* Dp   = (const float*)d_in[10];  // (IN_,)
    const float* dtb  = (const float*)d_in[11];  // (IN_,)
    float* out = (float*)d_out;                  // (B, L, DM)

    // workspace layout (256B aligned slabs)
    char* ws = (char*)d_ws;
    size_t off = 0;
    auto take = [&](size_t bytes) {
        size_t o = off;
        off = (off + bytes + 255) & ~(size_t)255;
        return o;
    };
    unsigned short* hs_bf = (unsigned short*)(ws + take((size_t)MTOT * DM * 2));
    unsigned short* wi_bf = (unsigned short*)(ws + take((size_t)2 * IN_ * DM * 2));
    unsigned short* wd_bf = (unsigned short*)(ws + take((size_t)IN_ * IN_ * 2));
    unsigned short* wo_bf = (unsigned short*)(ws + take((size_t)DM * IN_ * 2));
    float*          xzf   = (float*)         (ws + take((size_t)MTOT * 2 * IN_ * 4));
    float*          u32   = (float*)         (ws + take((size_t)MTOT * IN_ * 4));
    unsigned short* ubf   = (unsigned short*)(ws + take((size_t)MTOT * IN_ * 2));
    float*          delta = (float*)         (ws + take((size_t)MTOT * IN_ * 4));
    unsigned short* ybf   = (unsigned short*)(ws + take((size_t)MTOT * IN_ * 2));
    (void)ws_size; (void)n_in; (void)in_sizes; (void)out_size;

    // 1) bf16 casts
    {
        int n;
        n = MTOT * DM;
        cast_f32_bf16_kernel<<<(n + 255) / 256, 256, 0, stream>>>(hs, hs_bf, n);
        n = 2 * IN_ * DM;
        cast_f32_bf16_kernel<<<(n + 255) / 256, 256, 0, stream>>>(wi, wi_bf, n);
        n = IN_ * IN_;
        cast_f32_bf16_kernel<<<(n + 255) / 256, 256, 0, stream>>>(wd, wd_bf, n);
        n = DM * IN_;
        cast_f32_bf16_kernel<<<(n + 255) / 256, 256, 0, stream>>>(wo, wo_bf, n);
    }

    // 2) in_proj: xz (MTOT x 3072) = hs (MTOT x 768) * wi^T
    gemm_bf16_tn_kernel<<<dim3((2 * IN_) / 128, MTOT / 128), 256, 0, stream>>>(
        hs_bf, wi_bf, xzf, DM, 2 * IN_);

    // 3) causal depthwise conv + SiLU
    {
        int n = MTOT * IN_;
        conv_silu_kernel<<<(n + 255) / 256, 256, 0, stream>>>(xzf, cw, cb, u32, ubf);
    }

    // 4) dt_proj: delta (MTOT x 1536) = u * wd^T
    gemm_bf16_tn_kernel<<<dim3(IN_ / 128, MTOT / 128), 256, 0, stream>>>(
        ubf, wd_bf, delta, IN_, IN_);

    // 5) softplus
    {
        int n = MTOT * IN_;
        dt_softplus_kernel<<<(n + 255) / 256, 256, 0, stream>>>(delta, dpb, dtb);
    }

    // 6) selective scan + gate epilogue
    {
        int n = B_ * IN_ * NS;   // 49152
        scan_kernel<<<n / 256, 256, 0, stream>>>(delta, u32, xzf, alog, Bp, Cp, Dp, ybf);
    }

    // 7) out_proj: out (MTOT x 768) = y * wo^T
    gemm_bf16_tn_kernel<<<dim3(DM / 128, MTOT / 128), 256, 0, stream>>>(
        ybf, wo_bf, out, IN_, DM);
}
